// ContextBlock_69037304315907
// MI455X (gfx1250) — compile-verified, matrix operand order checked by reading, not run on hardware
//
#include <hip/hip_runtime.h>
#include <hip/hip_bf16.h>

typedef __attribute__((ext_vector_type(16))) _Float16 v16h;
typedef __attribute__((ext_vector_type(2)))  _Float16 h2;
typedef __attribute__((ext_vector_type(8)))  float    v8f;
typedef __attribute__((ext_vector_type(2)))  float    v2f;

#define B_   16
#define T_   2048
#define F_   64
#define A_   48
#define ROWS_PER_BLOCK 128
#define WIN_ROWS 176   // rows [t0-48 .. t0+127]

// fast sigmoid: prefer hardware v_tanh_f32 (1 trans op) over exp+rcp (2 trans ops)
__device__ __forceinline__ float sigmoidf_(float x) {
#if __has_builtin(__builtin_amdgcn_tanhf)
    return fmaf(0.5f, __builtin_amdgcn_tanhf(0.5f * x), 0.5f);
#else
    return __builtin_amdgcn_rcpf(1.0f + __expf(-x));
#endif
}

// v_cvt_pkrtz_f16_f32, bit_cast from the builtin's __fp16 vector type to _Float16 pair
__device__ __forceinline__ h2 pkrtz_(float a, float b) {
    return __builtin_bit_cast(h2, __builtin_amdgcn_cvt_pkrtz(a, b));
}

// K index inside a 32-wide K-half for f16 A/B fragments, per ISA layout:
// lanes 0-15 hold K = {0..7, 16..23}; lanes 16-31 hold K = {8..15, 24..31}
__device__ __forceinline__ int frag_k(int lane, int e) {
    return 8 * (lane >> 4) + e + ((e >= 8) ? 8 : 0);
}

struct H8 { h2 h[8]; };   // 16 f16 = one v16h worth of packed pairs

__global__ __launch_bounds__(256)
void ContextBlock_fused_kernel(const float* __restrict__ he,
                               const float* __restrict__ W1,
                               const float* __restrict__ W2,
                               float* __restrict__ out) {
    // Weights staged as WMMA A-fragments of W^T: [w][mtile][khalf][lane][e]
    __shared__ __align__(16) _Float16 lds_w[2 * 4 * 2 * 32 * 16];   // 16 KB
    __shared__ __align__(16) float    lds_win[WIN_ROWS * F_];       // 44 KB

    const int tid = threadIdx.x;
    const int b   = blockIdx.x >> 4;
    const int t0  = (blockIdx.x & 15) * ROWS_PER_BLOCK;
    const int win_base = t0 - A_;

    // ---- stage weight fragments: A[m][kk] = W^T[mtile*16+m][kh*32+kk] = W[row][col]
    for (int idx = tid; idx < 2 * 4 * 2 * 32 * 16; idx += 256) {
        int e  = idx & 15;
        int l  = (idx >> 4) & 31;
        int kh = (idx >> 9) & 1;
        int m  = (idx >> 10) & 3;
        int w  = idx >> 12;
        int kk  = frag_k(l, e);
        int row = kh * 32 + kk;
        int col = m * 16 + (l & 15);
        const float* wp = (w == 0) ? W1 : W2;
        lds_w[idx] = (_Float16)wp[row * F_ + col];
    }
    // ---- stage he window rows [t0-48 .. t0+127]
    for (int idx = tid; idx < WIN_ROWS * F_; idx += 256) {
        int s = idx >> 6;
        int f = idx & 63;
        int row = win_base + s;
        lds_win[idx] = (row >= 0) ? he[((size_t)b * T_ + row) * F_ + f] : 0.0f;
    }
    __syncthreads();

    const int wave = tid >> 5;
    const int lane = tid & 31;
    const int hi   = lane >> 4;
    const int n    = lane & 15;
    const int i    = t0 + wave * 16 + n;    // this lane's global t-row

    // ---- init B fragments of H0 = he row i (lane holds row n, K across elems)
    v16h Bh0, Bh1;
    {
        const float* hr = &lds_win[(48 + wave * 16 + n) * F_];
        #pragma unroll
        for (int e = 0; e < 16; ++e) {
            Bh0[e] = (_Float16)hr[frag_k(lane, e)];
            Bh1[e] = (_Float16)hr[32 + frag_k(lane, e)];
        }
    }

    const v16h* wfrag = (const v16h*)lds_w;  // fragment f at wfrag[f*32 + lane]
    const int L = min(A_, max(i, 1));

    float run_max = -3.0e38f;
    float denom   = 0.0f;
    v2f ctx2[16];
    #pragma unroll
    for (int q = 0; q < 16; ++q) ctx2[q] = (v2f){0.0f, 0.0f};

    for (int a = 0; a < A_; ++a) {
        // H = sigmoid(H @ W1), computed as W1^T tiles x H^T  (D layout == next B layout)
        v8f d0, d1, d2, d3;
        {
            v8f z = {};
            d0 = __builtin_amdgcn_wmma_f32_16x16x32_f16(false, wfrag[((0*4+0)*2+0)*32 + lane], false, Bh0, (short)0, z,  false, false);
            d0 = __builtin_amdgcn_wmma_f32_16x16x32_f16(false, wfrag[((0*4+0)*2+1)*32 + lane], false, Bh1, (short)0, d0, false, false);
            d1 = __builtin_amdgcn_wmma_f32_16x16x32_f16(false, wfrag[((0*4+1)*2+0)*32 + lane], false, Bh0, (short)0, z,  false, false);
            d1 = __builtin_amdgcn_wmma_f32_16x16x32_f16(false, wfrag[((0*4+1)*2+1)*32 + lane], false, Bh1, (short)0, d1, false, false);
            d2 = __builtin_amdgcn_wmma_f32_16x16x32_f16(false, wfrag[((0*4+2)*2+0)*32 + lane], false, Bh0, (short)0, z,  false, false);
            d2 = __builtin_amdgcn_wmma_f32_16x16x32_f16(false, wfrag[((0*4+2)*2+1)*32 + lane], false, Bh1, (short)0, d2, false, false);
            d3 = __builtin_amdgcn_wmma_f32_16x16x32_f16(false, wfrag[((0*4+3)*2+0)*32 + lane], false, Bh0, (short)0, z,  false, false);
            d3 = __builtin_amdgcn_wmma_f32_16x16x32_f16(false, wfrag[((0*4+3)*2+1)*32 + lane], false, Bh1, (short)0, d3, false, false);
        }
        // sigmoid + repack with v_cvt_pkrtz_f16_f32: D(mtile r) -> B(K-half elem)
        {
            H8 t0h, t1h;
            #pragma unroll
            for (int e = 0; e < 4; ++e) {
                t0h.h[e]     = pkrtz_(sigmoidf_(d0[2*e]), sigmoidf_(d0[2*e+1]));
                t0h.h[e + 4] = pkrtz_(sigmoidf_(d1[2*e]), sigmoidf_(d1[2*e+1]));
                t1h.h[e]     = pkrtz_(sigmoidf_(d2[2*e]), sigmoidf_(d2[2*e+1]));
                t1h.h[e + 4] = pkrtz_(sigmoidf_(d3[2*e]), sigmoidf_(d3[2*e+1]));
            }
            Bh0 = __builtin_bit_cast(v16h, t0h);
            Bh1 = __builtin_bit_cast(v16h, t1h);
        }
        // Y = sigmoid(H @ W2)
        {
            v8f z = {};
            d0 = __builtin_amdgcn_wmma_f32_16x16x32_f16(false, wfrag[((1*4+0)*2+0)*32 + lane], false, Bh0, (short)0, z,  false, false);
            d0 = __builtin_amdgcn_wmma_f32_16x16x32_f16(false, wfrag[((1*4+0)*2+1)*32 + lane], false, Bh1, (short)0, d0, false, false);
            d1 = __builtin_amdgcn_wmma_f32_16x16x32_f16(false, wfrag[((1*4+1)*2+0)*32 + lane], false, Bh0, (short)0, z,  false, false);
            d1 = __builtin_amdgcn_wmma_f32_16x16x32_f16(false, wfrag[((1*4+1)*2+1)*32 + lane], false, Bh1, (short)0, d1, false, false);
            d2 = __builtin_amdgcn_wmma_f32_16x16x32_f16(false, wfrag[((1*4+2)*2+0)*32 + lane], false, Bh0, (short)0, z,  false, false);
            d2 = __builtin_amdgcn_wmma_f32_16x16x32_f16(false, wfrag[((1*4+2)*2+1)*32 + lane], false, Bh1, (short)0, d2, false, false);
            d3 = __builtin_amdgcn_wmma_f32_16x16x32_f16(false, wfrag[((1*4+3)*2+0)*32 + lane], false, Bh0, (short)0, z,  false, false);
            d3 = __builtin_amdgcn_wmma_f32_16x16x32_f16(false, wfrag[((1*4+3)*2+1)*32 + lane], false, Bh1, (short)0, d3, false, false);
        }
        // gather he[b, j] from LDS window; dot with Y (this lane's 32 features)
        const int j = min(max(i - L + a, 0), T_ - 1);
        const float* g = &lds_win[(j - win_base) * F_];
        v2f heg2[16];
        #pragma unroll
        for (int m = 0; m < 4; ++m) {
            const float4* gp = (const float4*)&g[m * 16 + 8 * hi];
            float4 x0 = gp[0], x1 = gp[1];
            heg2[m*4+0] = (v2f){x0.x, x0.y};
            heg2[m*4+1] = (v2f){x0.z, x0.w};
            heg2[m*4+2] = (v2f){x1.x, x1.y};
            heg2[m*4+3] = (v2f){x1.z, x1.w};
        }
        v2f part2 = (v2f){0.0f, 0.0f};
        #pragma unroll
        for (int r = 0; r < 4; ++r) {
            v2f y0 = (v2f){sigmoidf_(d0[2*r]), sigmoidf_(d0[2*r+1])};
            v2f y1 = (v2f){sigmoidf_(d1[2*r]), sigmoidf_(d1[2*r+1])};
            v2f y2 = (v2f){sigmoidf_(d2[2*r]), sigmoidf_(d2[2*r+1])};
            v2f y3 = (v2f){sigmoidf_(d3[2*r]), sigmoidf_(d3[2*r+1])};
            part2 += y0 * heg2[r] + y1 * heg2[4 + r] + y2 * heg2[8 + r] + y3 * heg2[12 + r];
        }
        float part = part2[0] + part2[1];
        // lanes l and l^16 hold the same row n: SWAPX16 swizzle-add
        float other = __int_as_float(
            __builtin_amdgcn_ds_swizzle(__float_as_int(part), 0x401F));
        float s = part + other;
        s = (a < L) ? s : -1.0e9f;           // mask invalid steps like the reference

        // branchless online softmax + context accumulation
        float m_new = fmaxf(run_max, s);
        float scale = __expf(run_max - m_new);
        float p     = __expf(s - m_new);
        denom = denom * scale + p;
        v2f sc2 = (v2f){scale, scale};
        v2f p2  = (v2f){p, p};
        #pragma unroll
        for (int q = 0; q < 16; ++q) ctx2[q] = ctx2[q] * sc2 + p2 * heg2[q];
        run_max = m_new;
    }

    const float inv = __builtin_amdgcn_rcpf(denom);
    float* o = &out[((size_t)b * T_ + i) * F_ + 8 * hi];
    #pragma unroll
    for (int m = 0; m < 4; ++m) {
        float4 v0 = {ctx2[m*4+0][0]*inv, ctx2[m*4+0][1]*inv, ctx2[m*4+1][0]*inv, ctx2[m*4+1][1]*inv};
        float4 v1 = {ctx2[m*4+2][0]*inv, ctx2[m*4+2][1]*inv, ctx2[m*4+3][0]*inv, ctx2[m*4+3][1]*inv};
        *(float4*)&o[m * 16]     = v0;
        *(float4*)&o[m * 16 + 4] = v1;
    }
}

extern "C" void kernel_launch(void* const* d_in, const int* in_sizes, int n_in,
                              void* d_out, int out_size, void* d_ws, size_t ws_size,
                              hipStream_t stream) {
    const float* he = (const float*)d_in[0];   // (16, 2048, 64) f32
    const float* W1 = (const float*)d_in[1];   // (64, 64) f32
    const float* W2 = (const float*)d_in[2];   // (64, 64) f32
    // d_in[3] = attention_len (48), baked in at compile time
    float* out = (float*)d_out;                // (16, 2048, 64) f32

    dim3 grid(B_ * (T_ / ROWS_PER_BLOCK));     // 256 blocks
    dim3 block(256);                           // 8 waves x 16 rows
    ContextBlock_fused_kernel<<<grid, block, 0, stream>>>(he, W1, W2, out);
}
